// MetaGraphSelectiveLayer_38757784879146
// MI455X (gfx1250) — compile-verified, matrix operand order checked by reading, not run on hardware
//
#include <hip/hip_runtime.h>
#include <hip/hip_bf16.h>
#include <stdint.h>

// ---------------------------------------------------------------------------
// CDNA5 (gfx1250) WMMA types
// ---------------------------------------------------------------------------
typedef __attribute__((ext_vector_type(16))) __bf16 v16bf;
typedef __attribute__((ext_vector_type(8)))  float  v8f;

union FragBF { v16bf v; uint4 q[2]; };

#define CH   256
#define HID  256
#define HEADS 4
#define DK   64
#define MT   32   // GEMM row tile

// ---------------------------------------------------------------------------
// Weight transpose + f32 -> bf16 :  src[K][Nc] -> dst[Nc][K]
// ---------------------------------------------------------------------------
__global__ void wt_transpose_bf16(const float* __restrict__ src,
                                  __hip_bfloat16* __restrict__ dst,
                                  int K, int Nc) {
    int idx = blockIdx.x * 256 + threadIdx.x;
    if (idx >= K * Nc) return;
    int k = idx / Nc, n = idx % Nc;
    dst[(size_t)n * K + k] = __float2bfloat16(src[idx]);
}

// ---------------------------------------------------------------------------
// Edge prep: e_in = cat(x[row], x[col], edge_attr) [768] -> GN(32 groups of 24)
// -> ReLU -> bf16
// ---------------------------------------------------------------------------
__global__ __launch_bounds__(256)
void edge_prep_kernel(const float* __restrict__ x, const int* __restrict__ eidx,
                      const float* __restrict__ eattr,
                      const float* __restrict__ g1, const float* __restrict__ b1,
                      __hip_bfloat16* __restrict__ e_act, int E) {
    __shared__ float vals[3 * CH];
    __shared__ float mu[32], rs[32];
    int e = blockIdx.x;
    int t = threadIdx.x;
    int row = eidx[e];
    int col = eidx[E + e];
    vals[t]           = x[(size_t)row * CH + t];
    vals[CH + t]      = x[(size_t)col * CH + t];
    vals[2 * CH + t]  = eattr[(size_t)e * CH + t];
    __syncthreads();
    if (t < 32) {
        float s = 0.f, ss = 0.f;
#pragma unroll
        for (int i = 0; i < 24; i++) { float v = vals[t * 24 + i]; s += v; ss += v * v; }
        float m = s * (1.0f / 24.0f);
        float var = ss * (1.0f / 24.0f) - m * m;
        mu[t] = m;
        rs[t] = rsqrtf(var + 1e-5f);
    }
    __syncthreads();
#pragma unroll
    for (int j = 0; j < 3; j++) {
        int c = j * CH + t;
        int g = c / 24;
        float y = (vals[c] - mu[g]) * rs[g] * g1[c] + b1[c];
        e_act[(size_t)e * (3 * CH) + c] = __float2bfloat16(fmaxf(y, 0.f));
    }
}

// ---------------------------------------------------------------------------
// Node prep: x -> bf16 copy ; xa = ReLU(GN(x, 32 groups of 8)) -> cat_bf[:,0:256]
// ---------------------------------------------------------------------------
__global__ __launch_bounds__(256)
void node_prep_kernel(const float* __restrict__ x,
                      const float* __restrict__ g, const float* __restrict__ b,
                      __hip_bfloat16* __restrict__ x_bf,
                      __hip_bfloat16* __restrict__ cat_bf) {
    __shared__ float vals[CH];
    int n = blockIdx.x, t = threadIdx.x;
    float v = x[(size_t)n * CH + t];
    vals[t] = v;
    x_bf[(size_t)n * CH + t] = __float2bfloat16(v);
    __syncthreads();
    int gb = (t >> 3) << 3;
    float s = 0.f, ss = 0.f;
#pragma unroll
    for (int i = 0; i < 8; i++) { float w = vals[gb + i]; s += w; ss += w * w; }
    float m  = s * 0.125f;
    float rs = rsqrtf(ss * 0.125f - m * m + 1e-5f);
    float y  = fmaxf((v - m) * rs * g[t] + b[t], 0.f);
    cat_bf[(size_t)n * (2 * CH) + t] = __float2bfloat16(y);
}

// ---------------------------------------------------------------------------
// Shared GEMM core: 32 rows x 256 cols per 256-thread block (8 waves).
// Wave w owns columns [32w, 32w+32): 2 row-frags x 2 col-frags = 4 WMMA/k-step.
// Double-buffered A tile in LDS, one barrier per k-step, no guards in the
// hot loop (caller guarantees A rows padded to a multiple of 32).
// ---------------------------------------------------------------------------
#define GEMM_CORE(ACC)                                                          \
    int t = threadIdx.x;                                                        \
    int wave = t >> 5;                                                          \
    int lane = t & 31;                                                          \
    int rowBase = blockIdx.x * MT;                                              \
    int colBase = wave * 32;                                                    \
    v8f ACC[2][2];                                                              \
    _Pragma("unroll") for (int i = 0; i < 2; i++)                               \
        _Pragma("unroll") for (int j = 0; j < 2; j++)                           \
            ACC[i][j] = (v8f){0.f,0.f,0.f,0.f,0.f,0.f,0.f,0.f};                 \
    int arow = t >> 3;       /* 0..31 staging row  */                           \
    int aseg = t & 7;        /* 4 bf16 (8B) each   */                           \
    int frow = lane & 15;                                                       \
    int kb   = (lane < 16) ? 0 : 8;                                             \
    const __hip_bfloat16* arp = A + (size_t)(rowBase + arow) * K + aseg * 4;    \
    uint2 areg = *(const uint2*)(arp);                                          \
    *(uint2*)(&Atile[arow * 32 + aseg * 4]) = areg;                             \
    if (K > 32) areg = *(const uint2*)(arp + 32);                               \
    __syncthreads();                                                            \
    for (int k0 = 0; k0 < K; k0 += 32) {                                        \
        int cur = (k0 >> 5) & 1;                                                \
        FragBF a0, a1, b0, b1;                                                  \
        const __hip_bfloat16* ab = &Atile[cur * (MT * 32)];                     \
        a0.q[0] = *(const uint4*)(ab + frow * 32 + kb);                         \
        a0.q[1] = *(const uint4*)(ab + frow * 32 + kb + 16);                    \
        a1.q[0] = *(const uint4*)(ab + (16 + frow) * 32 + kb);                  \
        a1.q[1] = *(const uint4*)(ab + (16 + frow) * 32 + kb + 16);             \
        if (k0 + 32 < K)                                                        \
            *(uint2*)(&Atile[(1 - cur) * (MT * 32) + arow * 32 + aseg * 4]) = areg; \
        const uint4* bp0 = (const uint4*)(BT + (size_t)(colBase + frow)      * K + k0 + kb); \
        const uint4* bp1 = (const uint4*)(BT + (size_t)(colBase + 16 + frow) * K + k0 + kb); \
        b0.q[0] = bp0[0]; b0.q[1] = bp0[2];                                     \
        b1.q[0] = bp1[0]; b1.q[1] = bp1[2];                                     \
        ACC[0][0] = __builtin_amdgcn_wmma_f32_16x16x32_bf16(false, a0.v, false, b0.v, \
                                                            (short)0, ACC[0][0], false, false); \
        ACC[0][1] = __builtin_amdgcn_wmma_f32_16x16x32_bf16(false, a0.v, false, b1.v, \
                                                            (short)0, ACC[0][1], false, false); \
        ACC[1][0] = __builtin_amdgcn_wmma_f32_16x16x32_bf16(false, a1.v, false, b0.v, \
                                                            (short)0, ACC[1][0], false, false); \
        ACC[1][1] = __builtin_amdgcn_wmma_f32_16x16x32_bf16(false, a1.v, false, b1.v, \
                                                            (short)0, ACC[1][1], false, false); \
        __builtin_prefetch(arp + k0 + 64, 0, 0);                                \
        __syncthreads();                                                        \
        if (k0 + 64 < K) areg = *(const uint2*)(arp + k0 + 64);                 \
    }

// ---------------------------------------------------------------------------
// WMMA GEMM + bias + GroupNorm(32 groups of 8) + ReLU -> bf16
// ---------------------------------------------------------------------------
__global__ __launch_bounds__(256)
void gemm_gn_relu_kernel(const __hip_bfloat16* __restrict__ A,
                         const __hip_bfloat16* __restrict__ BT,
                         const float* __restrict__ bias,
                         const float* __restrict__ gamma,
                         const float* __restrict__ beta,
                         __hip_bfloat16* __restrict__ out,
                         int M, int K) {
    __shared__ __align__(16) __hip_bfloat16 Atile[2 * MT * 32];
    __shared__ float Ctile[MT * 256];
    GEMM_CORE(acc)
    // spill accumulators (+bias): VGPR r, lanes<16 -> M=r, lanes>=16 -> M=8+r
    {
        int mhi = (lane >> 4) * 8;
#pragma unroll
        for (int rt = 0; rt < 2; rt++) {
#pragma unroll
            for (int r = 0; r < 8; r++) {
                int mrow = rt * 16 + mhi + r;
                int cc0 = colBase + frow;
                int cc1 = colBase + 16 + frow;
                Ctile[mrow * 256 + cc0] = acc[rt][0][r] + bias[cc0];
                Ctile[mrow * 256 + cc1] = acc[rt][1][r] + bias[cc1];
            }
        }
    }
    __syncthreads();
    // GroupNorm(gs=8) + ReLU: thread handles 32 cols (4 groups) of one row
    int orow = t >> 3;
    int cb = (t & 7) * 32;
    int grow = rowBase + orow;
#pragma unroll
    for (int gblk = 0; gblk < 4; gblk++) {
        int gb = cb + gblk * 8;
        float s = 0.f, ss = 0.f;
#pragma unroll
        for (int i = 0; i < 8; i++) { float v = Ctile[orow * 256 + gb + i]; s += v; ss += v * v; }
        float m   = s * 0.125f;
        float rsg = rsqrtf(ss * 0.125f - m * m + 1e-5f);
#pragma unroll
        for (int i = 0; i < 8; i++) {
            int c = gb + i;
            float y = (Ctile[orow * 256 + c] - m) * rsg * gamma[c] + beta[c];
            if (grow < M)
                out[(size_t)grow * 256 + c] = __float2bfloat16(fmaxf(y, 0.f));
        }
    }
}

// ---------------------------------------------------------------------------
// WMMA GEMM + bias (+ optional f32 residual) -> optional f32 out / bf16 out
// ---------------------------------------------------------------------------
__global__ __launch_bounds__(256)
void gemm_bias_kernel(const __hip_bfloat16* __restrict__ A,
                      const __hip_bfloat16* __restrict__ BT,
                      const float* __restrict__ bias,
                      const float* __restrict__ residual,   // nullable, stride 256
                      float* __restrict__ outf,             // nullable, stride 256
                      __hip_bfloat16* __restrict__ outb,    // nullable
                      int outb_stride,
                      int M, int K) {
    __shared__ __align__(16) __hip_bfloat16 Atile[2 * MT * 32];
    GEMM_CORE(acc)
    int mhi = (lane >> 4) * 8;
#pragma unroll
    for (int rt = 0; rt < 2; rt++) {
#pragma unroll
        for (int r = 0; r < 8; r++) {
            int mrow = rowBase + rt * 16 + mhi + r;
            if (mrow >= M) continue;
#pragma unroll
            for (int ct = 0; ct < 2; ct++) {
                int cc = colBase + ct * 16 + frow;
                float v = acc[rt][ct][r] + bias[cc];
                if (residual) v += residual[(size_t)mrow * 256 + cc];
                if (outf) outf[(size_t)mrow * 256 + cc] = v;
                if (outb) outb[(size_t)mrow * outb_stride + cc] = __float2bfloat16(v);
            }
        }
    }
}

// ---------------------------------------------------------------------------
// CSR build + V mean + segment-softmax attention
// ---------------------------------------------------------------------------
__global__ void zero_u32_kernel(uint32_t* p, int n) {
    int i = blockIdx.x * 256 + threadIdx.x;
    if (i < n) p[i] = 0u;
}

__global__ void deg_kernel(const int* __restrict__ eidx, int* deg, int E) {
    int e = blockIdx.x * 256 + threadIdx.x;
    if (e < E) atomicAdd(&deg[eidx[E + e]], 1);
}

__global__ void scan_kernel(const int* __restrict__ deg, int* off, int* cur, int N) {
    if (blockIdx.x == 0 && threadIdx.x == 0) {
        int acc = 0;
        for (int i = 0; i < N; i++) { off[i] = acc; cur[i] = acc; acc += deg[i]; }
        off[N] = acc;
    }
}

__global__ void fill_kernel(const int* __restrict__ eidx, int* cur, int* elist, int E) {
    int e = blockIdx.x * 256 + threadIdx.x;
    if (e < E) {
        int p = atomicAdd(&cur[eidx[E + e]], 1);
        elist[p] = e;
    }
}

__global__ __launch_bounds__(256)
void vmean_kernel(const float* __restrict__ v, float* vmean, int E) {
    int t = threadIdx.x;
    int r0 = blockIdx.x * 64;
    int r1 = min(r0 + 64, E);
    float s = 0.f;
    for (int r = r0; r < r1; r++) s += v[(size_t)r * 256 + t];
    atomicAdd(&vmean[t], s);
}

// one block per node; wave = head; lane owns 2 of the 64 head channels
__global__ __launch_bounds__(128)
void attn_kernel(const float* __restrict__ q, const float* __restrict__ kbuf,
                 const float* __restrict__ vbuf, const int* __restrict__ offsets,
                 const int* __restrict__ elist, const float* __restrict__ vmean,
                 __hip_bfloat16* __restrict__ g_bf, int E) {
    int n = blockIdx.x;
    int h = threadIdx.x >> 5;
    int lane = threadIdx.x & 31;
    int c0 = h * DK + lane * 2;
    float q0 = q[(size_t)n * CH + c0];
    float q1 = q[(size_t)n * CH + c0 + 1];
    int beg = offsets[n], end = offsets[n + 1];
    float g0, g1;
    if (beg == end) {
        // reference: softmax over all -1e9 -> uniform over ALL E edges -> mean of v
        float inv = 1.0f / (float)E;
        g0 = vmean[c0] * inv;
        g1 = vmean[c0 + 1] * inv;
    } else {
        const float scale = 0.125f;  // 1/sqrt(64)
        float mx = -1e30f;
        for (int i = beg; i < end; i++) {
            int e = elist[i];
            float d = q0 * kbuf[(size_t)e * CH + c0] + q1 * kbuf[(size_t)e * CH + c0 + 1];
#pragma unroll
            for (int o = 16; o > 0; o >>= 1) d += __shfl_xor(d, o, 32);
            mx = fmaxf(mx, d * scale);
        }
        float denom = 0.f, a0 = 0.f, a1 = 0.f;
        for (int i = beg; i < end; i++) {
            int e = elist[i];
            float d = q0 * kbuf[(size_t)e * CH + c0] + q1 * kbuf[(size_t)e * CH + c0 + 1];
#pragma unroll
            for (int o = 16; o > 0; o >>= 1) d += __shfl_xor(d, o, 32);
            float p = __expf(d * scale - mx);
            denom += p;
            a0 += p * vbuf[(size_t)e * CH + c0];
            a1 += p * vbuf[(size_t)e * CH + c0 + 1];
        }
        float inv = 1.0f / denom;
        g0 = a0 * inv; g1 = a1 * inv;
    }
    g_bf[(size_t)n * CH + c0]     = __float2bfloat16(g0);
    g_bf[(size_t)n * CH + c0 + 1] = __float2bfloat16(g1);
}

// ---------------------------------------------------------------------------
// Host launcher
// ---------------------------------------------------------------------------
extern "C" void kernel_launch(void* const* d_in, const int* in_sizes, int n_in,
                              void* d_out, int out_size, void* d_ws, size_t ws_size,
                              hipStream_t stream) {
    const float* x      = (const float*)d_in[0];
    const int*   eidx   = (const int*)d_in[1];
    const float* eattr  = (const float*)d_in[2];
    const float* gn_e1_g = (const float*)d_in[3];
    const float* gn_e1_b = (const float*)d_in[4];
    const float* We1 = (const float*)d_in[5];
    const float* be1 = (const float*)d_in[6];
    const float* gn_e2_g = (const float*)d_in[7];
    const float* gn_e2_b = (const float*)d_in[8];
    const float* We2 = (const float*)d_in[9];
    const float* be2 = (const float*)d_in[10];
    const float* gn_n_g = (const float*)d_in[11];
    const float* gn_n_b = (const float*)d_in[12];
    const float* Wq = (const float*)d_in[13];
    const float* bq = (const float*)d_in[14];
    const float* Wk = (const float*)d_in[15];
    const float* bk = (const float*)d_in[16];
    const float* Wv = (const float*)d_in[17];
    const float* bv = (const float*)d_in[18];
    const float* Wo = (const float*)d_in[19];
    const float* bo = (const float*)d_in[20];
    const float* Wn1 = (const float*)d_in[21];
    const float* bn1 = (const float*)d_in[22];
    const float* gn_n2_g = (const float*)d_in[23];
    const float* gn_n2_b = (const float*)d_in[24];
    const float* Wn2 = (const float*)d_in[25];
    const float* bn2 = (const float*)d_in[26];

    const int N = in_sizes[0] / CH;         // 2000
    const int E = in_sizes[1] / 2;          // 16000 (multiple of 32)
    const int Npad = (N + MT - 1) & ~(MT - 1);  // node buffers padded: no guards in GEMM loop

    // ---- workspace bump allocator (256B aligned) ----
    char* wp = (char*)d_ws;
    auto alloc = [&](size_t bytes) -> void* {
        void* p = (void*)wp;
        wp += (bytes + 255) & ~(size_t)255;
        return p;
    };
    typedef __hip_bfloat16 bf;
    bf* We1T = (bf*)alloc((size_t)3 * CH * HID * 2);
    bf* We2T = (bf*)alloc((size_t)HID * CH * 2);
    bf* WqT  = (bf*)alloc((size_t)CH * CH * 2);
    bf* WkT  = (bf*)alloc((size_t)CH * CH * 2);
    bf* WvT  = (bf*)alloc((size_t)CH * CH * 2);
    bf* WoT  = (bf*)alloc((size_t)CH * CH * 2);
    bf* Wn1T = (bf*)alloc((size_t)2 * CH * HID * 2);
    bf* Wn2T = (bf*)alloc((size_t)HID * CH * 2);
    bf* e_act   = (bf*)alloc((size_t)E * 3 * CH * 2);
    bf* h1      = (bf*)alloc((size_t)E * HID * 2);
    bf* eout_bf = (bf*)alloc((size_t)E * CH * 2);
    bf* x_bf    = (bf*)alloc((size_t)Npad * CH * 2);
    bf* cat_bf  = (bf*)alloc((size_t)Npad * 2 * CH * 2);
    bf* hn_bf   = (bf*)alloc((size_t)Npad * HID * 2);
    bf* g_bf    = (bf*)alloc((size_t)Npad * CH * 2);
    float* q_buf = (float*)alloc((size_t)Npad * CH * 4);
    float* k_buf = (float*)alloc((size_t)E * CH * 4);
    float* v_buf = (float*)alloc((size_t)E * CH * 4);
    float* vmean = (float*)alloc(CH * 4);
    int* deg     = (int*)alloc((size_t)N * 4);
    int* offs    = (int*)alloc((size_t)(N + 1) * 4);
    int* cur     = (int*)alloc((size_t)N * 4);
    int* elist   = (int*)alloc((size_t)E * 4);

    float* n_out = (float*)d_out;                     // [N, CH]
    float* e_out = (float*)d_out + (size_t)N * CH;    // [E, CH]

    dim3 blk256(256), blk128(128);
    auto cdiv = [](int a, int b) { return (a + b - 1) / b; };

    // 1) weight transposes (f32 -> bf16, [K,256] -> [256,K])
    wt_transpose_bf16<<<cdiv(3 * CH * HID, 256), blk256, 0, stream>>>(We1, We1T, 3 * CH, HID);
    wt_transpose_bf16<<<cdiv(HID * CH, 256),     blk256, 0, stream>>>(We2, We2T, HID, CH);
    wt_transpose_bf16<<<cdiv(CH * CH, 256),      blk256, 0, stream>>>(Wq,  WqT,  CH, CH);
    wt_transpose_bf16<<<cdiv(CH * CH, 256),      blk256, 0, stream>>>(Wk,  WkT,  CH, CH);
    wt_transpose_bf16<<<cdiv(CH * CH, 256),      blk256, 0, stream>>>(Wv,  WvT,  CH, CH);
    wt_transpose_bf16<<<cdiv(CH * CH, 256),      blk256, 0, stream>>>(Wo,  WoT,  CH, CH);
    wt_transpose_bf16<<<cdiv(2 * CH * HID, 256), blk256, 0, stream>>>(Wn1, Wn1T, 2 * CH, HID);
    wt_transpose_bf16<<<cdiv(HID * CH, 256),     blk256, 0, stream>>>(Wn2, Wn2T, HID, CH);

    // 2) edge path
    edge_prep_kernel<<<E, blk256, 0, stream>>>(x, eidx, eattr, gn_e1_g, gn_e1_b, e_act, E);
    gemm_gn_relu_kernel<<<cdiv(E, MT), blk256, 0, stream>>>(
        e_act, We1T, be1, gn_e2_g, gn_e2_b, h1, E, 3 * CH);
    gemm_bias_kernel<<<cdiv(E, MT), blk256, 0, stream>>>(
        h1, We2T, be2, eattr, e_out, eout_bf, CH, E, HID);

    // 3) projections
    node_prep_kernel<<<N, blk256, 0, stream>>>(x, gn_n_g, gn_n_b, x_bf, cat_bf);
    gemm_bias_kernel<<<cdiv(N, MT), blk256, 0, stream>>>(
        x_bf, WqT, bq, nullptr, q_buf, nullptr, CH, N, CH);
    gemm_bias_kernel<<<cdiv(E, MT), blk256, 0, stream>>>(
        eout_bf, WkT, bk, nullptr, k_buf, nullptr, CH, E, CH);
    gemm_bias_kernel<<<cdiv(E, MT), blk256, 0, stream>>>(
        eout_bf, WvT, bv, nullptr, v_buf, nullptr, CH, E, CH);

    // 4) CSR by destination node + V mean + segment softmax attention
    zero_u32_kernel<<<cdiv(N, 256), blk256, 0, stream>>>((uint32_t*)deg, N);
    zero_u32_kernel<<<1, blk256, 0, stream>>>((uint32_t*)vmean, CH);
    deg_kernel<<<cdiv(E, 256), blk256, 0, stream>>>(eidx, deg, E);
    scan_kernel<<<1, 1, 0, stream>>>(deg, offs, cur, N);
    fill_kernel<<<cdiv(E, 256), blk256, 0, stream>>>(eidx, cur, elist, E);
    vmean_kernel<<<cdiv(E, 64), blk256, 0, stream>>>(v_buf, vmean, E);
    attn_kernel<<<N, blk128, 0, stream>>>(q_buf, k_buf, v_buf, offs, elist, vmean, g_bf, E);

    // 5) output projection into cat buffer second half, then node MLP
    gemm_bias_kernel<<<cdiv(N, MT), blk256, 0, stream>>>(
        g_bf, WoT, bo, nullptr, nullptr, cat_bf + CH, 2 * CH, N, CH);
    gemm_gn_relu_kernel<<<cdiv(N, MT), blk256, 0, stream>>>(
        cat_bf, Wn1T, bn1, gn_n2_g, gn_n2_b, hn_bf, N, 2 * CH);
    gemm_bias_kernel<<<cdiv(N, MT), blk256, 0, stream>>>(
        hn_bf, Wn2T, bn2, x, n_out, nullptr, CH, N, HID);
}
